// DSVTInputLayer_29231547416592
// MI455X (gfx1250) — matrix operand order repaired and checked by reading, not compiled, not run on hardware
//
#include <hip/hip_runtime.h>
#include <hip/hip_bf16.h>
#include <stdint.h>

typedef _Float16 half_t;
typedef __attribute__((ext_vector_type(16))) _Float16 v16h;
typedef __attribute__((ext_vector_type(8)))  _Float16 v8h;
typedef __attribute__((ext_vector_type(8)))  float    v8f;
typedef __attribute__((ext_vector_type(4)))  uint32_t v4u;
typedef __attribute__((ext_vector_type(8)))  uint32_t v8u;

#define D_MODEL   192
#define N_LAYERS  8
#define WNUM      12800   // 4 batches * 3200 windows/sample
#define MAXV      144     // 12*12*1 voxels per window
#define SETSZ     36
#define MAXSETS   4       // ceil(144/36)
#define ROWS_PB   128     // voxel rows per GEMM block (8 waves x 16)

// ---------------------------------------------------------------------------
// Pass 1: 5 scalar moments per shift (sum x, y, x^2, y^2, xy). BN stats of the
// affine h = loc@W1+b1 are derived analytically from these.
// ---------------------------------------------------------------------------
__global__ void k_stats(const int* __restrict__ coors, int n,
                        double* __restrict__ stats) {
  int v = blockIdx.x * blockDim.x + threadIdx.x;
  double s[10];
#pragma unroll
  for (int i = 0; i < 10; ++i) s[i] = 0.0;
  if (v < n) {
    int x = coors[4 * v + 3], y = coors[4 * v + 2];
    float lx0 = (float)(x % 12) - 6.0f;
    float ly0 = (float)(y % 12) - 6.0f;
    float lx1 = (float)((x + 6) % 12) - 6.0f;
    float ly1 = (float)((y + 6) % 12) - 6.0f;
    s[0] = lx0; s[1] = ly0;
    s[2] = (double)lx0 * lx0; s[3] = (double)ly0 * ly0; s[4] = (double)lx0 * ly0;
    s[5] = lx1; s[6] = ly1;
    s[7] = (double)lx1 * lx1; s[8] = (double)ly1 * ly1; s[9] = (double)lx1 * ly1;
  }
#pragma unroll
  for (int off = 16; off > 0; off >>= 1) {
#pragma unroll
    for (int i = 0; i < 10; ++i) s[i] += __shfl_down(s[i], off, 32);
  }
  if ((threadIdx.x & 31) == 0) {
#pragma unroll
    for (int i = 0; i < 10; ++i) atomicAdd(&stats[i], s[i]);
  }
}

// ---------------------------------------------------------------------------
// Pass 2: fold BN into per-channel affine coefficients:
// relu(h*scale + bias) = relu(x*A1[k] + y*A2[k] + A0[k])
// ---------------------------------------------------------------------------
__global__ void k_coef(const double* __restrict__ stats,
                       const float* __restrict__ W1, const float* __restrict__ b1,
                       const float* __restrict__ gamma, const float* __restrict__ beta,
                       int n, float* __restrict__ coef) {
  int l = blockIdx.x;        // 8 layers
  int c = threadIdx.x;       // 192 channels
  int s = l & 1;
  const double* st = stats + s * 5;
  double invn = 1.0 / (double)n;
  double mx = st[0] * invn, my = st[1] * invn;
  double vxx = st[2] * invn - mx * mx;
  double vyy = st[3] * invn - my * my;
  double vxy = st[4] * invn - mx * my;
  float a  = W1[l * 384 + c];          // W1[l][0][c]
  float bw = W1[l * 384 + 192 + c];    // W1[l][1][c]
  float bv = b1[l * 192 + c];
  double mu  = (double)a * mx + (double)bw * my + (double)bv;
  double var = (double)a * a * vxx + (double)bw * bw * vyy + 2.0 * (double)a * bw * vxy;
  float scale = (float)((double)gamma[l * 192 + c] / sqrt(var + 1e-5));
  float bias  = beta[l * 192 + c] - (float)mu * scale;
  coef[0 * 1536 + l * 192 + c] = a * scale;
  coef[1 * 1536 + l * 192 + c] = bw * scale;
  coef[2 * 1536 + l * 192 + c] = bv * scale + bias;
}

// ---------------------------------------------------------------------------
// W2 (8,192,192) f32 row-major [k][n]  ->  f16 transposed [n][k] so B-fragment
// K runs are contiguous vector loads.
// ---------------------------------------------------------------------------
__global__ void k_w2t(const float* __restrict__ W2, half_t* __restrict__ W2T) {
  int idx = blockIdx.x * blockDim.x + threadIdx.x;
  if (idx >= N_LAYERS * D_MODEL * D_MODEL) return;
  int l = idx / (D_MODEL * D_MODEL);
  int rem = idx % (D_MODEL * D_MODEL);
  int nn = rem / D_MODEL, kk = rem % D_MODEL;
  W2T[(size_t)l * D_MODEL * D_MODEL + (size_t)nn * D_MODEL + kk] =
      (half_t)W2[(size_t)l * D_MODEL * D_MODEL + (size_t)kk * D_MODEL + nn];
}

// ---------------------------------------------------------------------------
// Window id per voxel + per-window counts (both shifts in one pass).
// ---------------------------------------------------------------------------
__global__ void k_win(const int* __restrict__ coors, int n,
                      int* __restrict__ win0, int* __restrict__ win1,
                      int* __restrict__ cnt0, int* __restrict__ cnt1) {
  int v = blockIdx.x * blockDim.x + threadIdx.x;
  if (v >= n) return;
  int b = coors[4 * v], y = coors[4 * v + 2], x = coors[4 * v + 3];
  int w0 = b * 3200 + (x / 12) * 80 + (y / 12) * 2;
  int w1 = b * 3200 + ((x + 6) / 12) * 80 + ((y + 6) / 12) * 2;
  win0[v] = w0; win1[v] = w1;
  atomicAdd(&cnt0[w0], 1);
  atomicAdd(&cnt1[w1], 1);
}

// Serial prefix over window-id order (== sorted-unique order) -> set bases + S
__global__ void k_scan(const int* __restrict__ cnt, int* __restrict__ setBase,
                       int* __restrict__ S) {
  for (int s = 0; s < 2; ++s) {
    int acc = 0;
    const int* c = cnt + s * WNUM;
    int* sb = setBase + s * WNUM;
    for (int w = 0; w < WNUM; ++w) {
      sb[w] = acc;
      acc += (c[w] + SETSZ - 1) / SETSZ;
    }
    S[s] = acc;
  }
}

// Deterministic cell ownership: min voxel index per (window, key-ordered cell)
__global__ void k_cell(const int* __restrict__ coors, const int* __restrict__ win,
                       int n, int part, int sh, unsigned* __restrict__ owner) {
  int v = blockIdx.x * blockDim.x + threadIdx.x;
  if (v >= n) return;
  int y = coors[4 * v + 2], x = coors[4 * v + 3];
  int cx = (x + sh) % 12, cy = (y + sh) % 12;
  int cell = (part == 0) ? (cy * 12 + cx) : (cx * 12 + cy);
  atomicMin(&owner[(size_t)win[v] * MAXV + cell], (unsigned)v);
}

// In-place compaction of occupied cells -> key-sorted inner order (pad = -1)
__global__ void k_compact(unsigned* __restrict__ owner) {
  int w = blockIdx.x * blockDim.x + threadIdx.x;
  if (w >= WNUM) return;
  unsigned* row = owner + (size_t)w * MAXV;
  int r = 0;
  for (int c = 0; c < MAXV; ++c) {
    unsigned u = row[c];
    if (u != 0xFFFFFFFFu) row[r++] = u;
  }
  for (; r < MAXV; ++r) row[r] = 0xFFFFFFFFu;
}

// Emit inds + mask for one (shift, partition).
__global__ void k_write(const unsigned* __restrict__ owner,
                        const int* __restrict__ cnt, const int* __restrict__ setBase,
                        const int* __restrict__ S, int sidx, int part,
                        float* __restrict__ out, size_t featCount) {
  int idx = blockIdx.x * blockDim.x + threadIdx.x;
  if (idx >= WNUM * MAXSETS * SETSZ) return;
  int j  = idx % SETSZ;
  int ls = (idx / SETSZ) % MAXSETS;
  int w  = idx / (SETSZ * MAXSETS);
  int c  = cnt[w];
  int sn = (c + SETSZ - 1) / SETSZ;
  if (ls >= sn) return;
  int g = setBase[w] + ls;
  int base = ls * SETSZ + j;
  int den = sn * SETSZ;
  int sel = (base * c) / den;
  if (sel > MAXV - 1) sel = MAXV - 1;
  int val = (int)owner[(size_t)w * MAXV + sel];
  int prev;
  if (j == 0) {
    prev = -1;
  } else {
    int selp = ((base - 1) * c) / den;
    if (selp > MAXV - 1) selp = MAXV - 1;
    prev = (int)owner[(size_t)w * MAXV + selp];
  }
  float m = (val == prev) ? 1.0f : 0.0f;
  size_t S0 = (size_t)S[0];
  size_t Ss = (size_t)S[sidx];
  size_t indsBase = featCount + (sidx ? 144ull * S0 : 0ull);
  size_t o = (size_t)part * Ss * SETSZ + (size_t)g * SETSZ + (size_t)j;
  out[indsBase + o] = (float)val;                  // inds
  out[indsBase + 72ull * Ss + o] = m;              // mask
}

// ---------------------------------------------------------------------------
// Fused BN+ReLU+GEMM. Each block: 128 rows x 192 cols of one layer.
//  * wave 0 issues a TDM tensor_load_to_lds of the whole 192x192 f16 W2^T
//    layer (72 KB) into LDS; D# built per ISA section 8 (count=1, type=2,
//    data_size=2B, 1-D tile of 36864 elements).
//  * per-layer coefficient vectors staged with global_load_async_to_lds_b32.
//  * 8 waves each own a 16-row tile; B fragments are ds_load_b128 from LDS;
//    A fragments synthesized on the fly; 6x v_wmma_f32_16x16x32_f16 per tile.
// ---------------------------------------------------------------------------
__global__ __launch_bounds__(256) void k_gemm(
    const int* __restrict__ coors, int n,
    const float* __restrict__ coef, const float* __restrict__ b2,
    const half_t* __restrict__ W2T, float* __restrict__ pos_out) {
  __shared__ __attribute__((aligned(16))) half_t lw2[D_MODEL * D_MODEL];
  __shared__ __attribute__((aligned(16))) float cA1[D_MODEL], cA2[D_MODEL],
                                                cA0[D_MODEL], cB2[D_MODEL];
  __shared__ float xs[ROWS_PB], ys[ROWS_PB];
  int l = blockIdx.y;
  int blockRow = blockIdx.x * ROWS_PB;
  int t = threadIdx.x;

  // --- TDM stage of the B matrix (one descriptor, wave 0 only) ---
  if (t == 0) {
    uint64_t ga = (uint64_t)(uintptr_t)(W2T + (size_t)l * D_MODEL * D_MODEL);
    uint32_t la = (uint32_t)(uintptr_t)(&lw2[0]);
    const uint32_t NE = (uint32_t)(D_MODEL * D_MODEL);  // 36864 elements (2B)
    v4u g0;
    g0[0] = 1u;                                          // count=1, user flags 0
    g0[1] = la;                                          // lds_addr
    g0[2] = (uint32_t)ga;                                // global_addr[31:0]
    g0[3] = (uint32_t)((ga >> 32) & 0x1FFFFFFu) | (2u << 30); // addr[56:32]|type=2
    v8u g1;
    g1[0] = 1u << 16;                 // workgroup_mask=0, data_size=1 (2 bytes)
    g1[1] = (NE & 0xFFFFu) << 16;     // tensor_dim0[15:0] at bits 63:48
    g1[2] = (NE >> 16) | (1u << 16);  // tensor_dim0[31:16] | tensor_dim1=1
    g1[3] = (NE & 0xFFFFu) << 16;     // tile_dim0 at bits 127:112
    g1[4] = 1u;                       // tile_dim1=1, tile_dim2=0
    g1[5] = NE;                       // tensor_dim0_stride[31:0]
    g1[6] = 0u;
    g1[7] = 0u;
    asm volatile("tensor_load_to_lds %0, %1" :: "s"(g0), "s"(g1) : "memory");
  }

  // --- async stage of per-layer coefficient vectors ---
  if (t < D_MODEL) {
    asm volatile("global_load_async_to_lds_b32 %0, %1, off"
                 :: "v"((uint32_t)(uintptr_t)&cA1[t]),
                    "v"((uint64_t)(uintptr_t)&coef[0 * 1536 + l * D_MODEL + t])
                 : "memory");
    asm volatile("global_load_async_to_lds_b32 %0, %1, off"
                 :: "v"((uint32_t)(uintptr_t)&cA2[t]),
                    "v"((uint64_t)(uintptr_t)&coef[1 * 1536 + l * D_MODEL + t])
                 : "memory");
    asm volatile("global_load_async_to_lds_b32 %0, %1, off"
                 :: "v"((uint32_t)(uintptr_t)&cA0[t]),
                    "v"((uint64_t)(uintptr_t)&coef[2 * 1536 + l * D_MODEL + t])
                 : "memory");
    asm volatile("global_load_async_to_lds_b32 %0, %1, off"
                 :: "v"((uint32_t)(uintptr_t)&cB2[t]),
                    "v"((uint64_t)(uintptr_t)&b2[l * D_MODEL + t])
                 : "memory");
  }

  // locs for this block's 128 rows
  if (t < ROWS_PB) {
    int row = blockRow + t;
    if (row >= n) row = n - 1;
    int sh = (l & 1) ? 6 : 0;
    xs[t] = (float)((coors[4 * row + 3] + sh) % 12) - 6.0f;
    ys[t] = (float)((coors[4 * row + 2] + sh) % 12) - 6.0f;
  }

#if __has_builtin(__builtin_amdgcn_s_wait_tensorcnt)
  __builtin_amdgcn_s_wait_tensorcnt(0);
#else
  asm volatile("s_wait_tensorcnt 0x0" ::: "memory");
#endif
  asm volatile("s_wait_asynccnt 0x0" ::: "memory");
  __syncthreads();

  int wave = t >> 5;
  int lane = t & 31;
  int rowbase = blockRow + wave * 16;
  if (rowbase + 16 <= n) {            // wave-level guard; tiles are full
    int r  = lane & 15;
    int hi = lane >> 4;
    float xv = xs[wave * 16 + r], yv = ys[wave * 16 + r];
    for (int ct = 0; ct < 12; ++ct) {
      int col = ct * 16 + r;
      v8f acc = {0.f, 0.f, 0.f, 0.f, 0.f, 0.f, 0.f, 0.f};
#pragma unroll
      for (int kk = 0; kk < 6; ++kk) {
        int kbase = kk * 32;
        v16h a;
#pragma unroll
        for (int e = 0; e < 16; ++e) {
          // A 16-bit 16x32 layout: lanes 0-15 hold K {0..7,16..23}, 16-31 {8..15,24..31}
          int k = kbase + ((e < 8) ? (hi * 8 + e) : (16 + hi * 8 + (e - 8)));
          float hv = fmaf(xv, cA1[k], fmaf(yv, cA2[k], cA0[k]));
          a[e] = (half_t)fmaxf(hv, 0.0f);
        }
        // B 16-bit 32x16 layout from LDS: lanes 0-15 K {0..7,16..23}, 16-31 {8..15,24..31}
        const v8h* bp = (const v8h*)(lw2 + (size_t)col * D_MODEL + kbase + hi * 8);
        v8h b0 = bp[0];
        v8h b1v = bp[2];
        v16h b = __builtin_shufflevector(b0, b1v, 0, 1, 2, 3, 4, 5, 6, 7,
                                         8, 9, 10, 11, 12, 13, 14, 15);
        acc = __builtin_amdgcn_wmma_f32_16x16x32_f16(
            false, a, false, b, (short)0, acc, false, false);
      }
      float bb = cB2[col];
#pragma unroll
      for (int i = 0; i < 8; ++i) {
        int row = rowbase + hi * 8 + i;  // D layout: M = vgpr + 8*(lane>=16)
        pos_out[((size_t)l * n + row) * D_MODEL + col] = acc[i] + bb;
      }
    }
  }
}

// ---------------------------------------------------------------------------
extern "C" void kernel_launch(void* const* d_in, const int* in_sizes, int n_in,
                              void* d_out, int out_size, void* d_ws, size_t ws_size,
                              hipStream_t stream) {
  (void)n_in; (void)ws_size;
  const float* voxel_feats = (const float*)d_in[0];
  const int*   coors       = (const int*)d_in[1];
  const float* W1    = (const float*)d_in[2];
  const float* b1    = (const float*)d_in[3];
  const float* gamma = (const float*)d_in[4];
  const float* beta  = (const float*)d_in[5];
  const float* W2    = (const float*)d_in[6];
  const float* b2    = (const float*)d_in[7];
  float* out = (float*)d_out;
  int n = in_sizes[1] / 4;
  size_t featCount = (size_t)n * D_MODEL;

  // workspace layout
  char* ws = (char*)d_ws;
  double*  stats   = (double*)ws;                               // 10 doubles
  float*   coef    = (float*)(ws + 128);                        // 3*1536 f32
  half_t*  W2T     = (half_t*)(ws + 128 + 18432);               // 8*192*192 f16
  char*    p       = ws + 128 + 18432 + 589824;
  int*     win0    = (int*)p;            p += (size_t)n * 4;
  int*     win1    = (int*)p;            p += (size_t)n * 4;
  int*     cnt     = (int*)p;            p += 2 * WNUM * 4;
  int*     setBase = (int*)p;            p += 2 * WNUM * 4;
  int*     S       = (int*)p;            p += 64;
  unsigned* owner  = (unsigned*)p;       // WNUM*MAXV unsigned = 7.37 MB

  // 1) voxel_feats pass-through
  hipMemcpyAsync(out, voxel_feats, featCount * sizeof(float),
                 hipMemcpyDeviceToDevice, stream);

  // 2) pos-embed coefficient pipeline
  hipMemsetAsync(stats, 0, 10 * sizeof(double), stream);
  int nb = (n + 255) / 256;
  k_stats<<<nb, 256, 0, stream>>>(coors, n, stats);
  k_coef<<<N_LAYERS, D_MODEL, 0, stream>>>(stats, W1, b1, gamma, beta, n, coef);
  k_w2t<<<(N_LAYERS * D_MODEL * D_MODEL + 255) / 256, 256, 0, stream>>>(W2, W2T);

  // 3) window partitioning
  hipMemsetAsync(cnt, 0, 2 * WNUM * sizeof(int), stream);
  k_win<<<nb, 256, 0, stream>>>(coors, n, win0, win1, cnt, cnt + WNUM);
  k_scan<<<1, 1, 0, stream>>>(cnt, setBase, S);
  for (int s = 0; s < 2; ++s) {
    for (int part = 0; part < 2; ++part) {
      hipMemsetAsync(owner, 0xFF, (size_t)WNUM * MAXV * sizeof(unsigned), stream);
      k_cell<<<nb, 256, 0, stream>>>(coors, s ? win1 : win0, n, part, s ? 6 : 0, owner);
      k_compact<<<(WNUM + 255) / 256, 256, 0, stream>>>(owner);
      k_write<<<(WNUM * MAXSETS * SETSZ + 255) / 256, 256, 0, stream>>>(
          owner, cnt + s * WNUM, setBase + s * WNUM, S, s, part, out, featCount);
    }
  }

  // 4) fused BN+ReLU+GEMM -> pos_embeds (last 8*N*192 floats of d_out)
  float* pos_out = out + ((size_t)out_size - (size_t)N_LAYERS * featCount);
  dim3 ggrid((n + ROWS_PB - 1) / ROWS_PB, N_LAYERS);
  k_gemm<<<ggrid, 256, 0, stream>>>(coors, n, coef, b2, W2T, pos_out);
}
